// Model_49469433315659
// MI455X (gfx1250) — matrix operand order repaired and checked by reading, not compiled, not run on hardware
//
#include <hip/hip_runtime.h>

// ---------------------------------------------------------------------------
// Model: PatchTST-style forecaster with memory retrieval, MHA, gated fusion.
// B=16, L=512, NV=64, D=512, N_PATCH=64, MEM=8192, TOPK=8, NHEAD=4 (dh=128),
// PRED=192, HEAD_NF=32768.  All heavy GEMMs use v_wmma_f32_16x16x32_bf16,
// staged through LDS with GLOBAL_LOAD_ASYNC_TO_LDS_B128 + double buffering.
// ---------------------------------------------------------------------------

typedef unsigned short u16;
typedef __attribute__((ext_vector_type(4)))  unsigned int u32x4;
typedef __attribute__((ext_vector_type(16))) __bf16       v16bf;
typedef __attribute__((ext_vector_type(8)))  float        v8f;

union Frag { v16bf v; u32x4 u[2]; };

__device__ __forceinline__ u16 f2bf_bits(float f) {
    unsigned u = __builtin_bit_cast(unsigned, f);
    unsigned r = u + 0x7FFFu + ((u >> 16) & 1u);   // round-to-nearest-even
    return (u16)(r >> 16);
}
__device__ __forceinline__ float bf2f(u16 h) {
    unsigned u = ((unsigned)h) << 16;
    return __builtin_bit_cast(float, u);
}
__device__ __forceinline__ float gelu_exact(float x) {
    return 0.5f * x * (1.0f + erff(x * 0.70710678118654752f));
}
__device__ __forceinline__ v8f wmma_bf16(v16bf a, v16bf b, v8f c) {
    return __builtin_amdgcn_wmma_f32_16x16x32_bf16(false, a, false, b,
                                                   (short)0, c, false, false);
}

// LDS byte offset of a __shared__ pointer: generic LDS addresses put the
// wave-relative offset in the low 32 bits (flat->LDS truncation, ISA 10.2).
__device__ __forceinline__ unsigned lds_off32(const void* p) {
    return (unsigned)(size_t)p;
}
// Async DMA: 16 bytes global -> LDS, tracked by ASYNCcnt (ISA 10.7 / 15.18.3).
__device__ __forceinline__ void async_b128(const u16* gsrc, unsigned ldsdst) {
    asm volatile("global_load_async_to_lds_b128 %0, %1, off"
                 :: "v"(ldsdst), "v"(gsrc) : "memory");
}

// ---------------------------------------------------------------------------
// Instance-norm statistics over time axis.  grid=1024 (b*64+v), block=256.
// stats[0..1023]=mean, stats[1024..2047]=stdev = sqrt(var+1e-5)/0.4
// ---------------------------------------------------------------------------
__global__ __launch_bounds__(256)
void stats_kernel(const float* __restrict__ x, float* __restrict__ stats) {
    int bv = blockIdx.x, b = bv >> 6, v = bv & 63;
    __shared__ float s1[256], s2[256];
    float a = 0.f, q = 0.f;
    for (int l = threadIdx.x; l < 512; l += 256) {
        float t = x[((size_t)b * 512 + l) * 64 + v];
        a += t; q += t * t;
    }
    s1[threadIdx.x] = a; s2[threadIdx.x] = q;
    __syncthreads();
    for (int s = 128; s > 0; s >>= 1) {
        if (threadIdx.x < s) { s1[threadIdx.x] += s1[threadIdx.x + s];
                               s2[threadIdx.x] += s2[threadIdx.x + s]; }
        __syncthreads();
    }
    if (threadIdx.x == 0) {
        float mean = s1[0] * (1.0f / 512.0f);
        float var  = s2[0] * (1.0f / 512.0f) - mean * mean;
        stats[bv]        = mean;
        stats[1024 + bv] = sqrtf(var + 1e-5f) * 2.5f;   // /0.4
    }
}

// ---------------------------------------------------------------------------
// Patch embedding + sinusoidal positional encoding -> emb_bf [1024,64,512]
// grid=(1024,64), block=512.
// ---------------------------------------------------------------------------
__global__ __launch_bounds__(512)
void emb_kernel(const float* __restrict__ x_enc, const float* __restrict__ W_val,
                const float* __restrict__ stats, u16* __restrict__ emb_bf) {
    int bn = blockIdx.x, n = blockIdx.y;
    int b = bn >> 6, v = bn & 63;
    __shared__ float patch[16];
    float mean = stats[bn];
    float inv  = 1.0f / stats[1024 + bn];
    if (threadIdx.x < 16) {
        int t = n * 8 + threadIdx.x;
        if (t > 511) t = 511;                       // replication pad
        patch[threadIdx.x] = (x_enc[((size_t)b * 512 + t) * 64 + v] - mean) * inv;
    }
    __syncthreads();
    int d = threadIdx.x;
    float s = 0.f;
    #pragma unroll
    for (int j = 0; j < 16; ++j) s += patch[j] * W_val[j * 512 + d];
    float di   = (float)(d & ~1);
    float freq = expf(di * (-9.210340371976184f / 512.0f)); // -ln(1e4)/D
    float ang  = (float)n * freq;
    s += (d & 1) ? cosf(ang) : sinf(ang);
    emb_bf[((size_t)bn * 64 + n) * 512 + d] = f2bf_bits(s);
}

// qflat = mean over patches.  grid=1024, block=512.
__global__ __launch_bounds__(512)
void qflat_kernel(const u16* __restrict__ emb_bf, u16* __restrict__ qflat_bf) {
    int bn = blockIdx.x, d = threadIdx.x;
    float s = 0.f;
    for (int n = 0; n < 64; ++n) s += bf2f(emb_bf[((size_t)bn * 64 + n) * 512 + d]);
    qflat_bf[(size_t)bn * 512 + d] = f2bf_bits(s * (1.0f / 64.0f));
}

// generic fp32 -> bf16 converter
__global__ __launch_bounds__(256)
void cvt_kernel(const float* __restrict__ src, u16* __restrict__ dst, long long n) {
    long long i = (long long)blockIdx.x * 256 + threadIdx.x;
    if (i < n) dst[i] = f2bf_bits(src[i]);
}

// ---------------------------------------------------------------------------
// bf16 WMMA GEMM.  C[M,N] = act(A[M,K] @ B + bias [+ Cprev])
//   A row-major bf16 (lda=K).  B: TRANSB ? [N,K] : [K,N] bf16.
//   Block tile 128x64, BK=32, 256 threads = 8 waves (4x2), wave tile 32x32.
//   Double-buffered LDS; A (and B when TRANSB) staged with async-to-LDS DMA.
//   Requires M%128==0, N%64==0, K%32==0 (true for all call sites).
// ---------------------------------------------------------------------------
template <int TRANSB>
__global__ __launch_bounds__(256)
void gemm_bf16_kernel(const u16* __restrict__ A, const u16* __restrict__ Bm,
                      const float* __restrict__ bias,
                      float* __restrict__ Cf, u16* __restrict__ Ch,
                      int M, int N, int K, int accum, int act) {
    __shared__ __align__(16) u16 As[2][128][40];
    __shared__ __align__(16) u16 Bs[2][64][40];
    const int tid  = threadIdx.x;
    const int lane = tid & 31;
    const int wave = tid >> 5;
    const int wm   = wave >> 1, wn = wave & 1;
    const int g    = lane >> 4, hl = lane & 15;
    const long long m0 = (long long)blockIdx.y * 128;
    const long long n0 = (long long)blockIdx.x * 64;

    v8f acc[2][2];
    #pragma unroll
    for (int i = 0; i < 2; ++i)
        #pragma unroll
        for (int j = 0; j < 2; ++j)
            #pragma unroll
            for (int e = 0; e < 8; ++e) acc[i][j][e] = 0.0f;

    const int ar  = tid >> 1;           // A stage: 0..127
    const int ac  = (tid & 1) * 16;     // 0 / 16
    const int brT = tid >> 2;           // B stage (TRANSB): 0..63
    const int bcT = (tid & 3) * 8;      // 0,8,16,24
    const int bk  = tid >> 3;           // B stage (normal): 0..31
    const int bn2 = (tid & 7) * 8;      // 0..56

    auto stage = [&](int buf, int k0) {
        const u16* asrc = A + (m0 + ar) * (long long)K + k0 + ac;
        async_b128(asrc,     lds_off32(&As[buf][ar][ac]));
        async_b128(asrc + 8, lds_off32(&As[buf][ar][ac + 8]));
        if (TRANSB) {
            async_b128(Bm + (n0 + brT) * (long long)K + k0 + bcT,
                       lds_off32(&Bs[buf][brT][bcT]));
        } else {
            // transpose-scatter: 8 contiguous bf16 along N -> Bs[n][k]
            u32x4 t = *(const u32x4*)(Bm + (long long)(k0 + bk) * N + n0 + bn2);
            const u16* t16 = (const u16*)&t;
            #pragma unroll
            for (int e = 0; e < 8; ++e) Bs[buf][bn2 + e][bk] = t16[e];
        }
    };

    stage(0, 0);
    const int nIter = K >> 5;
    for (int it = 0; it < nIter; ++it) {
        const int cur = it & 1;
        if (it + 1 < nIter) {
            stage(cur ^ 1, (it + 1) << 5);
            // current buffer's async loads (issued earlier, in-order) are done
            // once only the next tile's loads remain outstanding.
            if (TRANSB) asm volatile("s_wait_asynccnt 0x3" ::: "memory");
            else        asm volatile("s_wait_asynccnt 0x2" ::: "memory");
        } else {
            asm volatile("s_wait_asynccnt 0x0" ::: "memory");
        }
        __syncthreads();

        Frag a[2], b[2];
        #pragma unroll
        for (int mi = 0; mi < 2; ++mi) {            // A frag: K = base_j+8g+h
            int r = wm * 32 + mi * 16 + hl;
            a[mi].u[0] = *(const u32x4*)&As[cur][r][8 * g];
            a[mi].u[1] = *(const u32x4*)&As[cur][r][16 + 8 * g];
        }
        #pragma unroll
        for (int ni = 0; ni < 2; ++ni) {            // B frag: K = base_j+8g+h
            int r = wn * 32 + ni * 16 + hl;
            b[ni].u[0] = *(const u32x4*)&Bs[cur][r][8 * g];
            b[ni].u[1] = *(const u32x4*)&Bs[cur][r][16 + 8 * g];
        }
        #pragma unroll
        for (int mi = 0; mi < 2; ++mi)
            #pragma unroll
            for (int ni = 0; ni < 2; ++ni)
                acc[mi][ni] = wmma_bf16(a[mi].v, b[ni].v, acc[mi][ni]);
        __syncthreads();   // all waves done reading `cur` before it is restaged
    }

    // epilogue: C/D layout -> row = e + 8g, col = hl
    #pragma unroll
    for (int mi = 0; mi < 2; ++mi)
        #pragma unroll
        for (int ni = 0; ni < 2; ++ni)
            #pragma unroll
            for (int e = 0; e < 8; ++e) {
                long long m = m0 + wm * 32 + mi * 16 + e + 8 * g;
                long long n = n0 + wn * 32 + ni * 16 + hl;
                float val = acc[mi][ni][e];
                if (bias)  val += bias[n];
                if (accum) val += Cf[m * N + n];
                if (act)   val = gelu_exact(val);
                if (Cf) Cf[m * N + n] = val;
                if (Ch) Ch[m * N + n] = f2bf_bits(val);
            }
}

// ---------------------------------------------------------------------------
// Top-8 per row of sim [1024, 8192] (destructive).  grid=1024, block=256.
// ---------------------------------------------------------------------------
__global__ __launch_bounds__(256)
void topk_kernel(float* __restrict__ sim, int* __restrict__ idx) {
    int row = blockIdx.x;
    float* s = sim + (size_t)row * 8192;
    __shared__ float bv[256];
    __shared__ int   bi[256];
    for (int sel = 0; sel < 8; ++sel) {
        float best = -3.0e38f; int besti = 0;
        for (int j = threadIdx.x; j < 8192; j += 256) {
            float t = s[j];
            if (t > best) { best = t; besti = j; }
        }
        bv[threadIdx.x] = best; bi[threadIdx.x] = besti;
        __syncthreads();
        for (int st = 128; st > 0; st >>= 1) {
            if (threadIdx.x < st) {
                float o = bv[threadIdx.x + st]; int oi = bi[threadIdx.x + st];
                if (o > bv[threadIdx.x] ||
                    (o == bv[threadIdx.x] && oi < bi[threadIdx.x])) {
                    bv[threadIdx.x] = o; bi[threadIdx.x] = oi;
                }
            }
            __syncthreads();
        }
        if (threadIdx.x == 0) { idx[row * 8 + sel] = bi[0]; s[bi[0]] = -3.0e38f; }
        __syncthreads();
    }
}

// gather retrieved memory rows -> bf16.  grid=8192 (bn*8+k), block=256.
__global__ __launch_bounds__(256)
void gather_kernel(const float* __restrict__ mem_bank, const int* __restrict__ idx,
                   u16* __restrict__ retr) {
    int rk = blockIdx.x;
    int src = idx[rk];
    for (int d = threadIdx.x; d < 512; d += 256)
        retr[(size_t)rk * 512 + d] = f2bf_bits(mem_bank[(size_t)src * 512 + d]);
}

// mean over TOPK of the memory MLP output.  grid=1024, block=512.
__global__ __launch_bounds__(512)
void mean8_kernel(const float* __restrict__ h2, float* __restrict__ localvec) {
    int bn = blockIdx.x, d = threadIdx.x;
    float s = 0.f;
    #pragma unroll
    for (int k = 0; k < 8; ++k) s += h2[((size_t)bn * 8 + k) * 512 + d];
    localvec[(size_t)bn * 512 + d] = s * 0.125f;
}

// local = localvec + emb, to bf16.  grid-stride over 65536*512.
__global__ __launch_bounds__(256)
void localbf_kernel(const float* __restrict__ localvec, const u16* __restrict__ emb,
                    u16* __restrict__ localb, long long total) {
    long long i = (long long)blockIdx.x * 256 + threadIdx.x;
    if (i >= total) return;
    long long m = i >> 9; int d = (int)(i & 511);
    long long bn = m >> 6;
    localb[i] = f2bf_bits(localvec[bn * 512 + d] + bf2f(emb[i]));
}

// ---------------------------------------------------------------------------
// Fused MHA per (bn, head).  One wave (32 threads, EXEC all ones).
// Q,K read straight from global bf16; V transposed into LDS; softmax in LDS.
// ---------------------------------------------------------------------------
__global__ __launch_bounds__(32)
void attn_kernel(const u16* __restrict__ q, const u16* __restrict__ k,
                 const u16* __restrict__ v, u16* __restrict__ o) {
    __shared__ __align__(16) u16   vsT[128][72];   // V transposed: [dh][key]
    __shared__ __align__(16) float Ss[64][68];     // scores / probs fp32
    __shared__ __align__(16) u16   Ps[64][72];     // probs bf16
    const int bn = blockIdx.x, h = blockIdx.y;
    const int lane = threadIdx.x;
    const int g = lane >> 4, hl = lane & 15;
    const size_t rowbase = (size_t)bn * 64;
    const int cbase = h * 128;

    // stage V transposed
    for (int key = lane; key < 64; key += 32) {
        const u16* src = v + (rowbase + key) * 512 + cbase;
        for (int d = 0; d < 128; d += 8) {
            u32x4 t = *(const u32x4*)(src + d);
            const u16* t16 = (const u16*)&t;
            #pragma unroll
            for (int e = 0; e < 8; ++e) vsT[d + e][key] = t16[e];
        }
    }
    __syncthreads();

    // S = Q K^T / sqrt(dh)
    const float scale = 0.08838834764831845f;      // 1/sqrt(128)
    for (int mi = 0; mi < 4; ++mi)
        for (int ni = 0; ni < 4; ++ni) {
            v8f acc;
            #pragma unroll
            for (int e = 0; e < 8; ++e) acc[e] = 0.f;
            #pragma unroll
            for (int t = 0; t < 4; ++t) {
                Frag a, b;
                const u16* ap = q + (rowbase + mi * 16 + hl) * 512 + cbase + t * 32;
                a.u[0] = *(const u32x4*)(ap + 8 * g);
                a.u[1] = *(const u32x4*)(ap + 16 + 8 * g);
                const u16* bp = k + (rowbase + ni * 16 + hl) * 512 + cbase + t * 32;
                b.u[0] = *(const u32x4*)(bp + 8 * g);
                b.u[1] = *(const u32x4*)(bp + 16 + 8 * g);
                acc = wmma_bf16(a.v, b.v, acc);
            }
            #pragma unroll
            for (int e = 0; e < 8; ++e)
                Ss[mi * 16 + e + 8 * g][ni * 16 + hl] = acc[e] * scale;
        }
    __syncthreads();

    // softmax rows -> Ps (bf16)
    for (int r = lane; r < 64; r += 32) {
        float mx = -1e30f;
        for (int c = 0; c < 64; ++c) mx = fmaxf(mx, Ss[r][c]);
        float sum = 0.f;
        for (int c = 0; c < 64; ++c) { float e = expf(Ss[r][c] - mx); sum += e; Ss[r][c] = e; }
        float inv = 1.0f / sum;
        for (int c = 0; c < 64; ++c) Ps[r][c] = f2bf_bits(Ss[r][c] * inv);
    }
    __syncthreads();

    // O = P V
    for (int j = 0; j < 8; ++j)
        for (int mi = 0; mi < 4; ++mi) {
            v8f acc;
            #pragma unroll
            for (int e = 0; e < 8; ++e) acc[e] = 0.f;
            #pragma unroll
            for (int t = 0; t < 2; ++t) {
                Frag a, b;
                const u16* ap = &Ps[mi * 16 + hl][t * 32];
                a.u[0] = *(const u32x4*)(ap + 8 * g);
                a.u[1] = *(const u32x4*)(ap + 16 + 8 * g);
                const u16* bp = &vsT[j * 16 + hl][t * 32];
                b.u[0] = *(const u32x4*)(bp + 8 * g);
                b.u[1] = *(const u32x4*)(bp + 16 + 8 * g);
                acc = wmma_bf16(a.v, b.v, acc);
            }
            #pragma unroll
            for (int e = 0; e < 8; ++e)
                o[(rowbase + mi * 16 + e + 8 * g) * 512 + cbase + j * 16 + hl] =
                    f2bf_bits(acc[e]);
        }
}

// gate = softmax over 2 of gelu(g1pre) @ W_g2 + b_g2.  grid=65536, block=128.
__global__ __launch_bounds__(128)
void gate2_kernel(const float* __restrict__ g1pre, const float* __restrict__ Wg2,
                  const float* __restrict__ bg2, float* __restrict__ gate) {
    int m = blockIdx.x;
    __shared__ float r0[128], r1[128];
    float a0 = 0.f, a1 = 0.f;
    for (int j = threadIdx.x; j < 512; j += 128) {
        float x = gelu_exact(g1pre[(size_t)m * 512 + j]);
        a0 += x * Wg2[j * 2];
        a1 += x * Wg2[j * 2 + 1];
    }
    r0[threadIdx.x] = a0; r1[threadIdx.x] = a1;
    __syncthreads();
    for (int s = 64; s > 0; s >>= 1) {
        if (threadIdx.x < s) { r0[threadIdx.x] += r0[threadIdx.x + s];
                               r1[threadIdx.x] += r1[threadIdx.x + s]; }
        __syncthreads();
    }
    if (threadIdx.x == 0) {
        float z0 = r0[0] + bg2[0], z1 = r1[0] + bg2[1];
        float mx = fmaxf(z0, z1);
        float e0 = expf(z0 - mx), e1 = expf(z1 - mx);
        float inv = 1.0f / (e0 + e1);
        gate[(size_t)m * 2]     = e0 * inv;
        gate[(size_t)m * 2 + 1] = e1 * inv;
    }
}

// flat = gate0*local + gate1*glob (bf16).  grid-stride over 65536*512.
__global__ __launch_bounds__(256)
void flatten_kernel(const float* __restrict__ gate, const u16* __restrict__ localb,
                    const u16* __restrict__ globb, u16* __restrict__ flat,
                    long long total) {
    long long i = (long long)blockIdx.x * 256 + threadIdx.x;
    if (i >= total) return;
    long long m = i >> 9;
    float g0 = gate[m * 2], g1 = gate[m * 2 + 1];
    flat[i] = f2bf_bits(g0 * bf2f(localb[i]) + g1 * bf2f(globb[i]));
}

// small fp32 FC: out = act(in @ W + b) [+ res]
__global__ __launch_bounds__(256)
void fc_kernel(const float* __restrict__ in, const float* __restrict__ W,
               const float* __restrict__ bias, const float* __restrict__ res,
               float* __restrict__ out, int M, int N, int K, int dogelu) {
    int i = blockIdx.x * 256 + threadIdx.x;
    if (i >= M * N) return;
    int m = i / N, n = i % N;
    float s = bias[n];
    for (int k = 0; k < K; ++k) s += in[(size_t)m * K + k] * W[(size_t)k * N + n];
    if (dogelu) s = gelu_exact(s);
    if (res) s += res[i];
    out[i] = s;
}

// out[b,p,v] = pred[b*64+v, p] * stdev + mean.  196608 elements.
__global__ __launch_bounds__(256)
void final_kernel(const float* __restrict__ pred, const float* __restrict__ stats,
                  float* __restrict__ out) {
    int i = blockIdx.x * 256 + threadIdx.x;
    if (i >= 16 * 192 * 64) return;
    int v = i & 63;
    int p = (i >> 6) % 192;
    int b = i / (192 * 64);
    int bv = b * 64 + v;
    out[i] = pred[(size_t)bv * 192 + p] * stats[1024 + bv] + stats[bv];
}

// ---------------------------------------------------------------------------
extern "C" void kernel_launch(void* const* d_in, const int* in_sizes, int n_in,
                              void* d_out, int out_size, void* d_ws, size_t ws_size,
                              hipStream_t stream) {
    (void)in_sizes; (void)n_in; (void)out_size; (void)ws_size;

    const float* x_enc    = (const float*)d_in[0];
    const float* W_val    = (const float*)d_in[1];
    const float* mem_bank = (const float*)d_in[2];
    const float* W_m1 = (const float*)d_in[3];  const float* b_m1 = (const float*)d_in[4];
    const float* W_m2 = (const float*)d_in[5];  const float* b_m2 = (const float*)d_in[6];
    const float* W_q  = (const float*)d_in[7];  const float* b_q  = (const float*)d_in[8];
    const float* W_k  = (const float*)d_in[9];  const float* b_k  = (const float*)d_in[10];
    const float* W_v  = (const float*)d_in[11]; const float* b_v  = (const float*)d_in[12];
    const float* W_o  = (const float*)d_in[13]; const float* b_o  = (const float*)d_in[14];
    const float* W_g1 = (const float*)d_in[15]; const float* b_g1 = (const float*)d_in[16];
    const float* W_g2 = (const float*)d_in[17]; const float* b_g2 = (const float*)d_in[18];
    const float* W_h  = (const float*)d_in[19]; const float* b_h  = (const float*)d_in[20];
    const float* W_f1 = (const float*)d_in[21]; const float* b_f1 = (const float*)d_in[22];
    const float* W_f2 = (const float*)d_in[23]; const float* b_f2 = (const float*)d_in[24];
    float* out = (float*)d_out;

    const long long MR = 65536;                  // BN * N_PATCH
    char* w = (char*)d_ws;
    size_t off = 0;
    auto alc = [&](size_t bytes) -> char* {
        char* p = w + off;
        off += (bytes + 255) & ~(size_t)255;
        return p;
    };

    float* stats    = (float*)alc(2048 * 4);
    u16* emb_bf     = (u16*)alc((size_t)MR * 512 * 2);
    u16* wq_bf      = (u16*)alc((size_t)512 * 512 * 2);
    u16* wk_bf      = (u16*)alc((size_t)512 * 512 * 2);
    u16* wv_bf      = (u16*)alc((size_t)512 * 512 * 2);
    u16* wo_bf      = (u16*)alc((size_t)512 * 512 * 2);
    u16* wm1_bf     = (u16*)alc((size_t)512 * 1024 * 2);
    u16* wm2_bf     = (u16*)alc((size_t)1024 * 512 * 2);
    u16* wg1_bf     = (u16*)alc((size_t)1024 * 512 * 2);
    u16* wh_bf      = (u16*)alc((size_t)32768 * 192 * 2);
    u16* mb_bf      = (u16*)alc((size_t)8192 * 512 * 2);
    u16* qflat_bf   = (u16*)alc((size_t)1024 * 512 * 2);
    float* scratch1 = (float*)alc((size_t)MR * 512 * 4);   // sim|h2, later g1pre
    float* sim      = scratch1;                             // 1024*8192
    float* h2       = scratch1 + (size_t)1024 * 8192;       // 8192*512
    float* g1pre    = scratch1;                             // reuse: 65536*512
    int*   idxbuf   = (int*)alc((size_t)8192 * 4);
    u16* retr_bf    = (u16*)alc((size_t)8192 * 512 * 2);
    u16* h1_bf      = (u16*)alc((size_t)8192 * 1024 * 2);
    float* localvec = (float*)alc((size_t)1024 * 512 * 4);
    u16* local_bf   = (u16*)alc((size_t)MR * 512 * 2);
    u16* q_bf       = (u16*)alc((size_t)MR * 512 * 2);
    u16* k_bf       = (u16*)alc((size_t)MR * 512 * 2);
    u16* v_bf       = (u16*)alc((size_t)MR * 512 * 2);
    u16* glob_pre   = (u16*)alc((size_t)MR * 512 * 2);
    u16* glob_bf    = k_bf;     // reuse after attention
    u16* flat_bf    = q_bf;     // reuse after attention
    float* gate     = (float*)alc((size_t)MR * 2 * 4);
    float* mh       = (float*)alc((size_t)1024 * 192 * 4);
    float* y1       = (float*)alc((size_t)1024 * 384 * 4);
    float* pred     = (float*)alc((size_t)1024 * 192 * 4);

    auto cvt = [&](const float* s, u16* d, size_t n) {
        cvt_kernel<<<(unsigned)((n + 255) / 256), 256, 0, stream>>>(s, d, (long long)n);
    };
    auto gemmN = [&](const u16* A, const u16* B, const float* bias, float* Cf,
                     u16* Ch, int M, int N, int K, int accum, int act) {
        dim3 grid((unsigned)(N / 64), (unsigned)(M / 128));
        gemm_bf16_kernel<0><<<grid, 256, 0, stream>>>(A, B, bias, Cf, Ch,
                                                      M, N, K, accum, act);
    };
    auto gemmT = [&](const u16* A, const u16* B, const float* bias, float* Cf,
                     u16* Ch, int M, int N, int K, int accum, int act) {
        dim3 grid((unsigned)(N / 64), (unsigned)(M / 128));
        gemm_bf16_kernel<1><<<grid, 256, 0, stream>>>(A, B, bias, Cf, Ch,
                                                      M, N, K, accum, act);
    };

    // 1. instance-norm stats
    stats_kernel<<<1024, 256, 0, stream>>>(x_enc, stats);

    // 2. weight / memory-bank conversions to bf16
    cvt(W_q, wq_bf, 512 * 512);     cvt(W_k, wk_bf, 512 * 512);
    cvt(W_v, wv_bf, 512 * 512);     cvt(W_o, wo_bf, 512 * 512);
    cvt(W_m1, wm1_bf, 512 * 1024);  cvt(W_m2, wm2_bf, 1024 * 512);
    cvt(W_g1, wg1_bf, 1024 * 512);  cvt(W_h, wh_bf, (size_t)32768 * 192);
    cvt(mem_bank, mb_bf, (size_t)8192 * 512);

    // 3. patch embedding (+pos emb), pooled query
    emb_kernel<<<dim3(1024, 64), 512, 0, stream>>>(x_enc, W_val, stats, emb_bf);
    qflat_kernel<<<1024, 512, 0, stream>>>(emb_bf, qflat_bf);

    // 4. similarity GEMM (B = mem_bank is [N,K] -> transposed B)
    gemmT(qflat_bf, mb_bf, nullptr, sim, nullptr, 1024, 8192, 512, 0, 0);

    // 5. top-k + gather + memory MLP + mean
    topk_kernel<<<1024, 256, 0, stream>>>(sim, idxbuf);
    gather_kernel<<<8192, 256, 0, stream>>>(mem_bank, idxbuf, retr_bf);
    gemmN(retr_bf, wm1_bf, b_m1, nullptr, h1_bf, 8192, 1024, 512, 0, 1); // gelu
    gemmN(h1_bf,  wm2_bf, b_m2, h2, nullptr,    8192, 512, 1024, 0, 0);
    mean8_kernel<<<1024, 512, 0, stream>>>(h2, localvec);
    localbf_kernel<<<(unsigned)((MR * 512 + 255) / 256), 256, 0, stream>>>(
        localvec, emb_bf, local_bf, MR * 512);

    // 6. attention: QKV projections, fused MHA, output projection
    gemmN(emb_bf, wq_bf, b_q, nullptr, q_bf, 65536, 512, 512, 0, 0);
    gemmN(emb_bf, wk_bf, b_k, nullptr, k_bf, 65536, 512, 512, 0, 0);
    gemmN(emb_bf, wv_bf, b_v, nullptr, v_bf, 65536, 512, 512, 0, 0);
    attn_kernel<<<dim3(1024, 4), 32, 0, stream>>>(q_bf, k_bf, v_bf, glob_pre);
    gemmN(glob_pre, wo_bf, b_o, nullptr, glob_bf, 65536, 512, 512, 0, 0);

    // 7. gated fusion: split-K gate GEMM (local half, then accumulate glob half)
    gemmN(local_bf, wg1_bf,             b_g1,    g1pre, nullptr, 65536, 512, 512, 0, 0);
    gemmN(glob_bf,  wg1_bf + 512 * 512, nullptr, g1pre, nullptr, 65536, 512, 512, 1, 0);
    gate2_kernel<<<65536, 128, 0, stream>>>(g1pre, W_g2, b_g2, gate);
    flatten_kernel<<<(unsigned)((MR * 512 + 255) / 256), 256, 0, stream>>>(
        gate, local_bf, glob_bf, flat_bf, MR * 512);

    // 8. memory head GEMM [1024,32768] @ [32768,192]
    gemmN(flat_bf, wh_bf, b_h, mh, nullptr, 1024, 192, 32768, 0, 0);

    // 9. fusion MLP + residual, denormalize
    fc_kernel<<<(1024 * 384 + 255) / 256, 256, 0, stream>>>(
        mh, W_f1, b_f1, nullptr, y1, 1024, 384, 192, 1);
    fc_kernel<<<(1024 * 192 + 255) / 256, 256, 0, stream>>>(
        y1, W_f2, b_f2, mh, pred, 1024, 192, 384, 1);
    final_kernel<<<(16 * 192 * 64 + 255) / 256, 256, 0, stream>>>(pred, stats, out);
}
